// TAttention_6914897346697
// MI455X (gfx1250) — compile-verified, matrix operand order checked by reading, not run on hardware
//
#include <hip/hip_runtime.h>

// ---------------------------------------------------------------------------
// Types / constants
// ---------------------------------------------------------------------------
typedef _Float16 half_t;
typedef __attribute__((ext_vector_type(16))) _Float16 v16h;
typedef __attribute__((ext_vector_type(8)))  _Float16 v8h;
typedef __attribute__((ext_vector_type(8)))  float    v8f;
typedef __attribute__((ext_vector_type(4)))  int      v4i;

constexpr int DM   = 512;     // model dim
constexpr int TT   = 512;     // sequence length
constexpr int NH   = 8;       // heads
constexpr int DH   = 64;      // head dim
constexpr long MTOT = 128L * 512L;   // B*T = 65536 rows

// ---------------------------------------------------------------------------
// Async global->LDS staging (gfx1250 GLOBAL_LOAD_ASYNC_TO_LDS_B128, ASYNCcnt)
// Builtin signature (from clang diagnostic): (int4 AS1*, int4 AS3*, imm, imm).
// Guarded by __has_builtin so a fall-back register path keeps compiles green.
// ---------------------------------------------------------------------------
#if __has_builtin(__builtin_amdgcn_global_load_async_to_lds_b128)
#define HAS_ASYNC_LDS 1
typedef __attribute__((address_space(1))) v4i* gptr16_t;
typedef __attribute__((address_space(3))) v4i* lptr16_t;
#else
#define HAS_ASYNC_LDS 0
#endif

__device__ __forceinline__ void cp16_async(half_t* ldst, const half_t* gsrc) {
#if HAS_ASYNC_LDS
  __builtin_amdgcn_global_load_async_to_lds_b128(
      (gptr16_t)(void*)gsrc, (lptr16_t)(void*)ldst, 0, 0);
#else
  *(v8h*)ldst = *(const v8h*)gsrc;
#endif
}

__device__ __forceinline__ void async_wait() {
#if HAS_ASYNC_LDS
#if __has_builtin(__builtin_amdgcn_s_wait_asynccnt)
  __builtin_amdgcn_s_wait_asynccnt(0);
#else
  asm volatile("s_wait_asynccnt 0x0" ::: "memory");
#endif
#endif
}

// ---------------------------------------------------------------------------
// WMMA helpers (CDNA5 16x16x32 f16 -> f32)
// A fragment: lane(0..15)=row M, half=lane>>4; v16h elems 0..7 = K half*8..+8,
// elems 8..15 = K 16+half*8..+8  (CDNA5 ISA 16-bit A 16x32 layout).
// B fragment: same striping with lane=column N, data must be K-contiguous,
// i.e. source laid out [N][K] (pre-transposed).
// C/D: VGPR r -> row r (+8 for lanes 16..31), col = lane&15.
// ---------------------------------------------------------------------------
__device__ __forceinline__ v16h ld_frag(const half_t* lds, int row, int stride,
                                        int koff, int half_id) {
  const half_t* p = lds + row * stride + koff + half_id * 8;
  v8h lo = *(const v8h*)(p);
  v8h hi = *(const v8h*)(p + 16);
  v16h r;
#pragma unroll
  for (int i = 0; i < 8; ++i) { r[i] = lo[i]; r[8 + i] = hi[i]; }
  return r;
}

__device__ __forceinline__ v8f wmma_f16(v16h a, v16h b, v8f c) {
  return __builtin_amdgcn_wmma_f32_16x16x32_f16(false, a, false, b,
                                                (short)0, c, false, false);
}

// ---------------------------------------------------------------------------
// LayerNorm (optionally fused residual add): one wave per 512-elem row.
// Writes f32 (residual path) and f16 (GEMM path) copies.
// ---------------------------------------------------------------------------
__global__ __launch_bounds__(256) void ln_fused(
    const float* __restrict__ x, const float* __restrict__ add,
    const float* __restrict__ gamma, const float* __restrict__ beta,
    float* __restrict__ out_f, half_t* __restrict__ out_h, int nrows) {
  const int wave = threadIdx.x >> 5, lane = threadIdx.x & 31;
  const long row = (long)blockIdx.x * 8 + wave;
  if (row >= nrows) return;
  const float* xr = x + row * DM;
  const float* ar = add ? add + row * DM : nullptr;
  float v[16], s = 0.f;
#pragma unroll
  for (int i = 0; i < 16; ++i) {
    int c = lane + i * 32;
    float t = xr[c];
    if (ar) t += ar[c];
    v[i] = t; s += t;
  }
#pragma unroll
  for (int m = 16; m >= 1; m >>= 1) s += __shfl_xor(s, m, 32);
  const float mu = s * (1.0f / DM);
  float var = 0.f;
#pragma unroll
  for (int i = 0; i < 16; ++i) { float d = v[i] - mu; var += d * d; }
#pragma unroll
  for (int m = 16; m >= 1; m >>= 1) var += __shfl_xor(var, m, 32);
  const float rstd = rsqrtf(var * (1.0f / DM) + 1e-5f);
#pragma unroll
  for (int i = 0; i < 16; ++i) {
    int c = lane + i * 32;
    float y = (v[i] - mu) * rstd * gamma[c] + beta[c];
    if (out_f) out_f[row * DM + c] = y;
    out_h[row * DM + c] = (half_t)y;
  }
}

// ---------------------------------------------------------------------------
// Weight convert + transpose: w[K][N] f32 -> wt[N][K] f16 (K-contiguous for B)
// ---------------------------------------------------------------------------
__global__ void wt_cvt(const float* __restrict__ w, half_t* __restrict__ wt) {
  int k = blockIdx.x;
  for (int n = threadIdx.x; n < DM; n += blockDim.x)
    wt[(size_t)n * DM + k] = (half_t)w[(size_t)k * DM + n];
}

// ---------------------------------------------------------------------------
// GEMM: C[M,512] = A[M,512] @ W  (W given as Wt[N][K] f16, pre-transposed)
// Block tile 128x128, 8 waves, wave = 32 rows x 64 cols (2x4 tiles,
// 8 accumulators -> 8 WMMAs per 6 fragment loads per K-step).
// Fused: +bias, ReLU, +f32 residual; f16 and/or f32 output.
// ---------------------------------------------------------------------------
constexpr int BM = 128, BN = 128, LDT = 40;  // 40-half row pad (80B, 16B ok)

__global__ __launch_bounds__(256) void gemm_wmma(
    const half_t* __restrict__ A, const half_t* __restrict__ Wt,
    const float* __restrict__ bias, const float* __restrict__ resid,
    half_t* __restrict__ out_h, float* __restrict__ out_f,
    int M, int relu) {
  __shared__ half_t sA[BM * LDT];   // 10 KB
  __shared__ half_t sB[BN * LDT];   // 10 KB
  const int tid = threadIdx.x;
  const int wave = tid >> 5, lane = tid & 31;
  const int half_id = lane >> 4, l16 = lane & 15;
  const int rg = wave & 3;          // row group: 32 rows each
  const int cg = wave >> 2;         // col group: 64 cols each
  const long m0 = (long)blockIdx.x * BM;
  const int n0 = blockIdx.y * BN;

  v8f acc[2][4] = {};
  for (int k0 = 0; k0 < DM; k0 += 32) {
    __syncthreads();
    {  // A tile 128x32 + B tile 128x32, async staged (2 b128 each per thread)
      int r = tid >> 1, off = (tid & 1) * 16;
      const half_t* asrc = A + (m0 + r) * DM + k0 + off;
      cp16_async(&sA[r * LDT + off],     asrc);
      cp16_async(&sA[r * LDT + off + 8], asrc + 8);
      const half_t* bsrc = Wt + (size_t)(n0 + r) * DM + k0 + off;
      cp16_async(&sB[r * LDT + off],     bsrc);
      cp16_async(&sB[r * LDT + off + 8], bsrc + 8);
    }
    async_wait();
    __syncthreads();
    v16h a0 = ld_frag(sA, rg * 32 + l16,      LDT, 0, half_id);
    v16h a1 = ld_frag(sA, rg * 32 + 16 + l16, LDT, 0, half_id);
#pragma unroll
    for (int c = 0; c < 4; ++c) {
      v16h b = ld_frag(sB, cg * 64 + c * 16 + l16, LDT, 0, half_id);
      acc[0][c] = wmma_f16(a0, b, acc[0][c]);
      acc[1][c] = wmma_f16(a1, b, acc[1][c]);
    }
  }
  // epilogue (C/D layout: VGPR r -> row r (+8 upper half), col = l16)
#pragma unroll
  for (int rt = 0; rt < 2; ++rt) {
#pragma unroll
    for (int c = 0; c < 4; ++c) {
      int col = n0 + cg * 64 + c * 16 + l16;
      float bv = bias ? bias[col] : 0.f;
#pragma unroll
      for (int r = 0; r < 8; ++r) {
        long gm = m0 + rg * 32 + rt * 16 + r + half_id * 8;
        float v = acc[rt][c][r] + bv;
        if (relu) v = v > 0.f ? v : 0.f;
        if (resid) v += resid[gm * DM + col];
        if (out_h) out_h[gm * DM + col] = (half_t)v;
        if (out_f) out_f[gm * DM + col] = v;
      }
    }
  }
}

// ---------------------------------------------------------------------------
// Attention: block = (b, h, 32-query-row tile). Whole 32x512 score strip kept
// in LDS (f32), exact softmax, then P(f16) @ V. ~110 KB dyn LDS (<320KB WGP).
// No 1/sqrt(dh) scaling, matching the reference.
// ---------------------------------------------------------------------------
constexpr int QR = 32;                 // query rows per block
constexpr int QPAD = 72;               // Q/K/V tile row pad (halves)
constexpr int SPAD = 516;              // score strip pad (floats)
constexpr int PPAD = 520;              // P strip pad (halves)
constexpr size_t SM_Q  = 0;
constexpr size_t SM_KV = SM_Q  + (size_t)QR * QPAD * 2;        // 4608
constexpr size_t SM_S  = SM_KV + (size_t)64 * QPAD * 2;        // 13824
constexpr size_t SM_P  = SM_S  + (size_t)QR * SPAD * 4;        // 79872
constexpr size_t ATTN_SMEM = SM_P + (size_t)QR * PPAD * 2;     // 113152 B

__global__ __launch_bounds__(256) void attn_wmma(
    const half_t* __restrict__ q, const half_t* __restrict__ k,
    const half_t* __restrict__ v, float* __restrict__ att) {
  extern __shared__ char smem[];
  half_t* sQ  = (half_t*)(smem + SM_Q);
  half_t* sKV = (half_t*)(smem + SM_KV);
  float*  sS  = (float*)(smem + SM_S);
  half_t* sP  = (half_t*)(smem + SM_P);

  const int tid = threadIdx.x;
  const int wave = tid >> 5, lane = tid & 31;
  const int half_id = lane >> 4, l16 = lane & 15;
  const int qt = blockIdx.x & 15;
  const int h  = (blockIdx.x >> 4) & 7;
  const int b  = blockIdx.x >> 7;
  const size_t base = (size_t)b * TT * DM + (size_t)h * DH;   // +row*DM
  const int rtile = wave >> 2, ctile = wave & 3;

  {  // load Q 32x64 (one b128 per thread, async)
    int r = tid >> 3, off = (tid & 7) * 8;
    cp16_async(&sQ[r * QPAD + off],
               q + base + (size_t)(qt * QR + r) * DM + off);
  }

  // ---- phase 1: S = Q @ K^T, 64 keys per LDS stage -----------------------
  for (int kb = 0; kb < 8; ++kb) {
    __syncthreads();
    {  // K block 64x64, row-major (d contiguous == K-contiguous for B frag)
      int r = tid >> 2, off = (tid & 3) * 16;
      const half_t* src = k + base + (size_t)(kb * 64 + r) * DM + off;
      cp16_async(&sKV[r * QPAD + off],     src);
      cp16_async(&sKV[r * QPAD + off + 8], src + 8);
    }
    async_wait();
    __syncthreads();
    v8f acc = {};
#pragma unroll
    for (int ks = 0; ks < 2; ++ks) {
      v16h a  = ld_frag(sQ,  rtile * 16 + l16, QPAD, ks * 32, half_id);
      v16h bf = ld_frag(sKV, ctile * 16 + l16, QPAD, ks * 32, half_id);
      acc = wmma_f16(a, bf, acc);
    }
#pragma unroll
    for (int r = 0; r < 8; ++r)
      sS[(rtile * 16 + r + half_id * 8) * SPAD + kb * 64 + ctile * 16 + l16] =
          acc[r];
  }
  __syncthreads();

  // ---- phase 2: softmax (8 lanes per row over the 32x512 strip) ----------
  {
    int row = tid >> 3, i = tid & 7;
    float* Sr = &sS[row * SPAD];
    float mx = -1e30f;
    for (int c = i * 64; c < i * 64 + 64; ++c) mx = fmaxf(mx, Sr[c]);
#pragma unroll
    for (int m = 4; m >= 1; m >>= 1) mx = fmaxf(mx, __shfl_xor(mx, m, 32));
    float sum = 0.f;
    for (int c = i * 64; c < i * 64 + 64; ++c) {
      float e = __expf(Sr[c] - mx);
      Sr[c] = e; sum += e;
    }
#pragma unroll
    for (int m = 4; m >= 1; m >>= 1) sum += __shfl_xor(sum, m, 32);
    float inv = 1.f / sum;
    half_t* Pr = &sP[row * PPAD];
    for (int c = i * 64; c < i * 64 + 64; ++c) Pr[c] = (half_t)(Sr[c] * inv);
  }

  // ---- phase 3: O = P @ V (V transposed during staging) ------------------
  v8f oacc = {};
  for (int kb = 0; kb < 8; ++kb) {
    __syncthreads();
    {  // Vt[d][key] so B-fragments are key-contiguous (VALU transpose)
      int key = tid >> 2, doff = (tid & 3) * 16;
      const half_t* src = v + base + (size_t)(kb * 64 + key) * DM + doff;
#pragma unroll
      for (int dd = 0; dd < 16; ++dd)
        sKV[(doff + dd) * QPAD + key] = src[dd];
    }
    __syncthreads();
#pragma unroll
    for (int ks = 0; ks < 2; ++ks) {
      v16h a  = ld_frag(sP,  rtile * 16 + l16, PPAD, kb * 64 + ks * 32, half_id);
      v16h bf = ld_frag(sKV, ctile * 16 + l16, QPAD, ks * 32, half_id);
      oacc = wmma_f16(a, bf, oacc);
    }
  }
#pragma unroll
  for (int r = 0; r < 8; ++r) {
    long grow = (long)b * TT + qt * QR + rtile * 16 + r + half_id * 8;
    att[grow * DM + h * DH + ctile * 16 + l16] = oacc[r];
  }
}

// ---------------------------------------------------------------------------
// Orchestration
// ---------------------------------------------------------------------------
extern "C" void kernel_launch(void* const* d_in, const int* in_sizes, int n_in,
                              void* d_out, int out_size, void* d_ws, size_t ws_size,
                              hipStream_t stream) {
  const float* x   = (const float*)d_in[0];
  const float* wq  = (const float*)d_in[1];
  const float* wk  = (const float*)d_in[2];
  const float* wv  = (const float*)d_in[3];
  const float* g1  = (const float*)d_in[4];
  const float* be1 = (const float*)d_in[5];
  const float* g2  = (const float*)d_in[6];
  const float* be2 = (const float*)d_in[7];
  const float* w1  = (const float*)d_in[8];
  const float* bf1 = (const float*)d_in[9];
  const float* w2  = (const float*)d_in[10];
  const float* bf2 = (const float*)d_in[11];
  float* out = (float*)d_out;

  // workspace carve-up
  char* ws = (char*)d_ws;
  const size_t F = (size_t)MTOT * DM * 4, Hs = (size_t)MTOT * DM * 2;
  float*  xn   = (float*)ws;  ws += F;    // LN1 out (f32, residual)
  float*  attb = (float*)ws;  ws += F;    // attention out (f32)
  float*  xt   = (float*)ws;  ws += F;    // LN2 out (f32, residual)
  half_t* xnh  = (half_t*)ws; ws += Hs;   // LN1 out f16
  half_t* qh   = (half_t*)ws; ws += Hs;
  half_t* kh   = (half_t*)ws; ws += Hs;
  half_t* vh   = (half_t*)ws; ws += Hs;
  half_t* xth  = (half_t*)ws; ws += Hs;   // LN2 out f16
  half_t* wqt  = (half_t*)ws; ws += (size_t)DM * DM * 2;
  half_t* wkt  = (half_t*)ws; ws += (size_t)DM * DM * 2;
  half_t* wvt  = (half_t*)ws; ws += (size_t)DM * DM * 2;
  half_t* w1t  = (half_t*)ws; ws += (size_t)DM * DM * 2;
  half_t* w2t  = (half_t*)ws; ws += (size_t)DM * DM * 2;
  half_t* h1   = qh;                       // alias: q dead after attention

  // weights -> f16, transposed [N][K]
  wt_cvt<<<DM, 256, 0, stream>>>(wq, wqt);
  wt_cvt<<<DM, 256, 0, stream>>>(wk, wkt);
  wt_cvt<<<DM, 256, 0, stream>>>(wv, wvt);
  wt_cvt<<<DM, 256, 0, stream>>>(w1, w1t);
  wt_cvt<<<DM, 256, 0, stream>>>(w2, w2t);

  // LN1
  ln_fused<<<MTOT / 8, 256, 0, stream>>>(x, nullptr, g1, be1, xn, xnh, (int)MTOT);

  // QKV projections
  dim3 gg(MTOT / BM, DM / BN);
  gemm_wmma<<<gg, 256, 0, stream>>>(xnh, wqt, nullptr, nullptr, qh, nullptr, (int)MTOT, 0);
  gemm_wmma<<<gg, 256, 0, stream>>>(xnh, wkt, nullptr, nullptr, kh, nullptr, (int)MTOT, 0);
  gemm_wmma<<<gg, 256, 0, stream>>>(xnh, wvt, nullptr, nullptr, vh, nullptr, (int)MTOT, 0);

  // attention (B * H * T/QR blocks)
  (void)hipFuncSetAttribute((const void*)attn_wmma,
                            hipFuncAttributeMaxDynamicSharedMemorySize,
                            (int)ATTN_SMEM);
  attn_wmma<<<128 * NH * (TT / QR), 256, ATTN_SMEM, stream>>>(qh, kh, vh, attb);

  // LN2 over (xn + att)
  ln_fused<<<MTOT / 8, 256, 0, stream>>>(xn, attb, g2, be2, xt, xth, (int)MTOT);

  // FFN: h1 = relu(xt@w1+bf1); out = xt + (h1@w2+bf2)
  gemm_wmma<<<gg, 256, 0, stream>>>(xth, w1t, bf1, nullptr, h1, nullptr, (int)MTOT, 1);
  gemm_wmma<<<gg, 256, 0, stream>>>(h1, w2t, bf2, xt, nullptr, out, (int)MTOT, 0);
}
